// AutoCompleteDecoderModel_64020782514682
// MI455X (gfx1250) — matrix operand order repaired and checked by reading, not compiled, not run on hardware
//
#include <hip/hip_runtime.h>
#include <hip/hip_bf16.h>
#include <math.h>

// ---------------- CDNA5 WMMA plumbing (wave32, 16x16x32 bf16 -> f32) ----------

typedef __attribute__((ext_vector_type(16))) __bf16 bf16x16;
typedef __attribute__((ext_vector_type(8))) float floatx8;

union Frag {
  bf16x16 v;
  unsigned int u[8];
};
union U8bf {
  uint4 u;
  __bf16 b[8];
};

__device__ inline float u2f(unsigned int x) {
  union { unsigned int u; float f; } c;
  c.u = x;
  return c.f;
}

__device__ inline floatx8 wmma_bf16(const Frag& a, const Frag& b, floatx8 c) {
  // (neg_a, A, neg_b, B, c_mod, C, reuse_a, reuse_b)
  return __builtin_amdgcn_wmma_f32_16x16x32_bf16(false, a.v, false, b.v, (short)0, c,
                                                 false, false);
}

// A fragment: 16x32 bf16 tile, row-major in LDS (row stride 32).
// ISA layout: lane = hi*16 + m ; VGPR v<4 -> K = 2v + 8*hi ; v>=4 -> K = 16 + 2(v-4) + 8*hi
__device__ inline void load_fragA(Frag& f, const __bf16* tile, int lane) {
  const int m = lane & 15, hi = lane >> 4;
  const __bf16* row = tile + m * 32;
#pragma unroll
  for (int v = 0; v < 8; ++v) {
    const int k = (v < 4) ? (2 * v + 8 * hi) : (16 + 2 * (v - 4) + 8 * hi);
    f.u[v] = *(const unsigned int*)(row + k);
  }
}

// B fragment: 32(K) x 16(N) tile stored as 16 N-rows x 32 K (i.e. W[n][k] row-major).
// ISA layout: lane = hi*16 + n ; VGPR v<4 -> K = 2v + 8*hi ; v>=4 -> K = 16 + 2(v-4) + 8*hi
__device__ inline void load_fragB(Frag& f, const __bf16* tile, int lane) {
  const int n = lane & 15, hi = lane >> 4;
  const __bf16* row = tile + n * 32;
#pragma unroll
  for (int v = 0; v < 8; ++v) {
    const int k = (v < 4) ? (2 * v + 8 * hi) : (16 + 2 * (v - 4) + 8 * hi);
    f.u[v] = *(const unsigned int*)(row + k);
  }
}

__device__ inline float sigmoidf_(float x) { return 1.f / (1.f + __expf(-x)); }

// Staging registers for one 16-element (32B) A slice: raw bits.
// f32 source -> 4 uint4 of float bits ; bf16 source -> 2 uint4.
struct AStage {
  uint4 q[4];
};

__device__ inline void stageA(AStage& st, const float* s0, long s0_stride, int s0w,
                              const __bf16* s1, int s1w, const __bf16* s2, int s2w,
                              int k0, int row, int c0) {
  if (k0 < s0w) {
    const uint4* src = (const uint4*)(s0 + (long)row * s0_stride + k0 + c0);
#pragma unroll
    for (int i = 0; i < 4; ++i) st.q[i] = src[i];
  } else if (k0 < s0w + s1w) {
    const uint4* src = (const uint4*)(s1 + (long)row * s1w + (k0 - s0w) + c0);
    st.q[0] = src[0];
    st.q[1] = src[1];
  } else {
    const uint4* src = (const uint4*)(s2 + (long)row * s2w + (k0 - s0w - s1w) + c0);
    st.q[0] = src[0];
    st.q[1] = src[1];
  }
}

// Commit staged A slice to LDS; f32 path converts to bf16 here (so the
// s_wait_loadcnt lands after the previous chunk's WMMAs, not before).
__device__ inline void commitA(const AStage& st, bool isF32, __bf16* dst) {
  if (isF32) {
#pragma unroll
    for (int i = 0; i < 4; ++i) {
      const uint4 t = st.q[i];
      dst[4 * i + 0] = (__bf16)u2f(t.x);
      dst[4 * i + 1] = (__bf16)u2f(t.y);
      dst[4 * i + 2] = (__bf16)u2f(t.z);
      dst[4 * i + 3] = (__bf16)u2f(t.w);
    }
  } else {
    *(uint4*)dst = st.q[0];
    *(uint4*)(dst + 8) = st.q[1];
  }
}

// ---------------- fused LSTM step (gates GEMM + elementwise) ------------------
// gates[128 x 2048] = [seg0(f32) | seg1(bf16) | seg2(bf16)] @ W^T + bih + bhh
// W is pre-packed bf16 [2048][K] with K = s0w + s1w + s2w.
// grid = 16 blocks (each owns 32 hidden cols x 4 gates), 256 threads = 8 waves.
// Double-buffered LDS pipeline: global loads for chunk k+1 overlap WMMAs of chunk k.
__global__ void __launch_bounds__(256)
lstm_step_kernel(const float* __restrict__ s0, long s0_stride, int s0w,
                 const __bf16* __restrict__ s1, int s1w,
                 const __bf16* __restrict__ s2, int s2w,
                 int K, const __bf16* __restrict__ W,
                 const float* __restrict__ bih, const float* __restrict__ bhh,
                 float* __restrict__ cbuf, __bf16* __restrict__ h_out,
                 __bf16* __restrict__ ehs_out, long ehs_stride) {
  __shared__ __align__(16) __bf16 sA[2][128][32];
  __shared__ __align__(16) __bf16 sB[2][4][32][32];
  const int tid = threadIdx.x, lane = tid & 31, w = tid >> 5;
  const int n0 = blockIdx.x * 32;
  const int rowA = tid >> 1, cA = (tid & 1) * 16;
  // two 16B B-slices staged per thread: 4 gates x 32 n-rows x 32 k = 512 chunks
  const int browa = tid >> 2, pA_ = (tid & 3) * 8;
  const int gatea = browa >> 5, na = browa & 31;
  const int browb = (tid + 256) >> 2, pB_ = ((tid + 256) & 3) * 8;
  const int gateb = browb >> 5, nb = browb & 31;
  const __bf16* Wa = W + (long)(gatea * 512 + n0 + na) * K + pA_;
  const __bf16* Wb = W + (long)(gateb * 512 + n0 + nb) * K + pB_;

  floatx8 acc[8] = {};  // [gate*2 + jsub]

  AStage pa;
  uint4 pb0, pb1;
  int pk = 0;
  stageA(pa, s0, s0_stride, s0w, s1, s1w, s2, s2w, 0, rowA, cA);
  pb0 = *(const uint4*)(Wa + 0);
  pb1 = *(const uint4*)(Wb + 0);

  int cur = 0;
  for (int k0 = 0; k0 < K; k0 += 32) {
    // commit staged chunk into lds[cur]
    commitA(pa, pk < s0w, &sA[cur][rowA][cA]);
    *(uint4*)&sB[cur][gatea][na][pA_] = pb0;
    *(uint4*)&sB[cur][gateb][nb][pB_] = pb1;
    __syncthreads();

    // issue global loads for next chunk (wait deferred to next commit)
    const int kn = k0 + 32;
    if (kn < K) {
      pk = kn;
      stageA(pa, s0, s0_stride, s0w, s1, s1w, s2, s2w, kn, rowA, cA);
      pb0 = *(const uint4*)(Wa + kn);
      pb1 = *(const uint4*)(Wb + kn);
    }

    // compute on lds[cur]
    Frag a;
    load_fragA(a, &sA[cur][w * 16][0], lane);
    Frag bf[8];
#pragma unroll
    for (int g = 0; g < 4; ++g)
#pragma unroll
      for (int j = 0; j < 2; ++j) load_fragB(bf[g * 2 + j], &sB[cur][g][j * 16][0], lane);
#pragma unroll
    for (int idx = 0; idx < 8; ++idx) acc[idx] = wmma_bf16(a, bf[idx], acc[idx]);
    __syncthreads();
    cur ^= 1;
  }

  // ---- elementwise LSTM cell (i,f,g,o live in matching acc slots) ----------
  const int hi = lane >> 4, l16 = lane & 15;
#pragma unroll
  for (int j = 0; j < 2; ++j) {
    const int colh = n0 + j * 16 + l16;
    const float bi = bih[colh] + bhh[colh];
    const float bf_ = bih[512 + colh] + bhh[512 + colh];
    const float bg = bih[1024 + colh] + bhh[1024 + colh];
    const float bo = bih[1536 + colh] + bhh[1536 + colh];
#pragma unroll
    for (int r = 0; r < 8; ++r) {
      const int rowb = w * 16 + r + 8 * hi;
      const long off = (long)rowb * 512 + colh;
      const float gi = acc[0 * 2 + j][r] + bi;
      const float gf = acc[1 * 2 + j][r] + bf_;
      const float gg = acc[2 * 2 + j][r] + bg;
      const float go = acc[3 * 2 + j][r] + bo;
      const float cn = sigmoidf_(gf) * cbuf[off] + sigmoidf_(gi) * tanhf(gg);
      cbuf[off] = cn;
      const float hn = sigmoidf_(go) * tanhf(cn);
      h_out[off] = (__bf16)hn;
      if (ehs_out) ehs_out[(long)rowb * ehs_stride + colh] = (__bf16)hn;
    }
  }
}

// ---------------- generic 128xN bf16 GEMM with epilogue options --------------
// out = A[128 x K] @ W^T (+bias). grid = N/32 blocks, 256 threads.
// outF: f32 store ; outB: raw bf16 store ; outT: tanh bf16 store (any may be null)
__global__ void __launch_bounds__(256)
gemm_kernel(const __bf16* __restrict__ A, int K, const __bf16* __restrict__ W, int Ntot,
            const float* __restrict__ bias, float* __restrict__ outF,
            __bf16* __restrict__ outB, __bf16* __restrict__ outT) {
  __shared__ __align__(16) __bf16 sA[2][128][32];
  __shared__ __align__(16) __bf16 sB[2][32][32];
  const int tid = threadIdx.x, lane = tid & 31, w = tid >> 5;
  const int n0 = blockIdx.x * 32;
  const int rowA = tid >> 1, cA = (tid & 1) * 16;
  const int browB = tid >> 2, pB_ = (tid & 3) * 8;  // only tid < 128 stages B
  const __bf16* Asrc = A + (long)rowA * K + cA;
  const __bf16* Wsrc = W + (long)(n0 + browB) * K + pB_;

  floatx8 acc[2] = {};

  uint4 a0, a1, b0 = {};
  a0 = *(const uint4*)(Asrc + 0);
  a1 = *(const uint4*)(Asrc + 8);
  if (tid < 128) b0 = *(const uint4*)(Wsrc + 0);

  int cur = 0;
  for (int k0 = 0; k0 < K; k0 += 32) {
    *(uint4*)&sA[cur][rowA][cA] = a0;
    *(uint4*)&sA[cur][rowA][cA + 8] = a1;
    if (tid < 128) *(uint4*)&sB[cur][browB][pB_] = b0;
    __syncthreads();

    const int kn = k0 + 32;
    if (kn < K) {
      a0 = *(const uint4*)(Asrc + kn);
      a1 = *(const uint4*)(Asrc + kn + 8);
      if (tid < 128) b0 = *(const uint4*)(Wsrc + kn);
    }

    Frag a;
    load_fragA(a, &sA[cur][w * 16][0], lane);
    Frag bf[2];
#pragma unroll
    for (int j = 0; j < 2; ++j) load_fragB(bf[j], &sB[cur][j * 16][0], lane);
#pragma unroll
    for (int j = 0; j < 2; ++j) acc[j] = wmma_bf16(a, bf[j], acc[j]);
    __syncthreads();
    cur ^= 1;
  }

  const int hi = lane >> 4, l16 = lane & 15;
#pragma unroll
  for (int j = 0; j < 2; ++j) {
    const int col = n0 + j * 16 + l16;
    const float bv = bias ? bias[col] : 0.f;
#pragma unroll
    for (int r = 0; r < 8; ++r) {
      const int row = w * 16 + r + 8 * hi;
      const float val = acc[j][r] + bv;
      const long off = (long)row * Ntot + col;
      if (outF) outF[off] = val;
      if (outB) outB[off] = (__bf16)val;
      if (outT) outT[off] = (__bf16)tanhf(val);
    }
  }
}

// ---------------- attention: scores -> masked softmax -> context -------------
// one block per batch row, 256 threads. enc_hs is bf16 [B][Lc][H] (L2-resident).
__global__ void __launch_bounds__(256)
attn_kernel(const __bf16* __restrict__ ehs, const float* __restrict__ q,
            const int* __restrict__ cpad, const __bf16* __restrict__ hdec,
            __bf16* __restrict__ hattn) {
  const int b = blockIdx.x, tid = threadIdx.x;
  __shared__ float sc[512];
  __shared__ float red[256];
  const __bf16* Eb = ehs + (long)b * 512 * 512;
  const float* qb = q + (long)b * 512;

  // pass 1: scores[l] = enc_hs[b,l,:] . q[b,:]  (masked)
  for (int l = tid; l < 512; l += 256) {
    const __bf16* row = Eb + (long)l * 512;
    float dot = 0.f;
    for (int h = 0; h < 512; h += 8) {
      U8bf u;
      u.u = *(const uint4*)(row + h);
      const float4 qa = *(const float4*)(qb + h);
      const float4 qc = *(const float4*)(qb + h + 4);
      dot += (float)u.b[0] * qa.x + (float)u.b[1] * qa.y + (float)u.b[2] * qa.z +
             (float)u.b[3] * qa.w + (float)u.b[4] * qc.x + (float)u.b[5] * qc.y +
             (float)u.b[6] * qc.z + (float)u.b[7] * qc.w;
    }
    sc[l] = (cpad[(long)b * 512 + l] != 0) ? -INFINITY : dot;
  }
  __syncthreads();

  // block max
  const float v0 = sc[tid], v1 = sc[tid + 256];
  red[tid] = fmaxf(v0, v1);
  __syncthreads();
  for (int s = 128; s > 0; s >>= 1) {
    if (tid < s) red[tid] = fmaxf(red[tid], red[tid + s]);
    __syncthreads();
  }
  const float mx = red[0];
  __syncthreads();

  const float e0 = __expf(v0 - mx), e1 = __expf(v1 - mx);
  sc[tid] = e0;
  sc[tid + 256] = e1;
  red[tid] = e0 + e1;
  __syncthreads();
  for (int s = 128; s > 0; s >>= 1) {
    if (tid < s) red[tid] += red[tid + s];
    __syncthreads();
  }
  const float inv = 1.f / red[0];
  __syncthreads();

  // pass 2: attn[h] = sum_l d[l] * enc_hs[b,l,h]
  float a0 = 0.f, a1 = 0.f;
  for (int l = 0; l < 512; ++l) {
    const float dl = sc[l];
    a0 += dl * (float)Eb[(long)l * 512 + tid];
    a1 += dl * (float)Eb[(long)l * 512 + tid + 256];
  }
  hattn[(long)b * 1024 + 512 + tid] = (__bf16)(a0 * inv);
  hattn[(long)b * 1024 + 512 + tid + 256] = (__bf16)(a1 * inv);
  // concat h into hattn[:, 0:512]
  hattn[(long)b * 1024 + tid] = hdec[(long)b * 512 + tid];
  hattn[(long)b * 1024 + tid + 256] = hdec[(long)b * 512 + tid + 256];
}

// ---------------- vocab GEMM + log-softmax + masked NLL ----------------------
// logits[128 x 128] = tanhV @ voc_W^T + voc_b ; single block, 256 threads.
__global__ void __launch_bounds__(256)
vocab_ce_kernel(const __bf16* __restrict__ A, const __bf16* __restrict__ W,
                const float* __restrict__ vb, const int* __restrict__ E, int t,
                float* __restrict__ nll_out, float* __restrict__ cnt_out) {
  __shared__ __align__(16) __bf16 sA[2][128][32];
  __shared__ __align__(16) __bf16 sB[2][128][32];
  __shared__ float red[256];
  const int tid = threadIdx.x, lane = tid & 31, w = tid >> 5;
  const int rowA = tid >> 1, cA = (tid & 1) * 16;
  const __bf16* Asrc = A + (long)rowA * 512 + cA;
  const __bf16* Wsrc = W + (long)rowA * 512 + cA;

  floatx8 acc[8] = {};

  uint4 a0 = *(const uint4*)(Asrc + 0), a1 = *(const uint4*)(Asrc + 8);
  uint4 b0 = *(const uint4*)(Wsrc + 0), b1 = *(const uint4*)(Wsrc + 8);

  int cur = 0;
  for (int k0 = 0; k0 < 512; k0 += 32) {
    *(uint4*)&sA[cur][rowA][cA] = a0;
    *(uint4*)&sA[cur][rowA][cA + 8] = a1;
    *(uint4*)&sB[cur][rowA][cA] = b0;
    *(uint4*)&sB[cur][rowA][cA + 8] = b1;
    __syncthreads();

    const int kn = k0 + 32;
    if (kn < 512) {
      a0 = *(const uint4*)(Asrc + kn);
      a1 = *(const uint4*)(Asrc + kn + 8);
      b0 = *(const uint4*)(Wsrc + kn);
      b1 = *(const uint4*)(Wsrc + kn + 8);
    }

    Frag a;
    load_fragA(a, &sA[cur][w * 16][0], lane);
    Frag bf[8];
#pragma unroll
    for (int j = 0; j < 8; ++j) load_fragB(bf[j], &sB[cur][j * 16][0], lane);
#pragma unroll
    for (int j = 0; j < 8; ++j) acc[j] = wmma_bf16(a, bf[j], acc[j]);
    __syncthreads();
    cur ^= 1;
  }

  const int hi = lane >> 4, l16 = lane & 15;
  float bcol[8];
#pragma unroll
  for (int j = 0; j < 8; ++j) bcol[j] = vb[j * 16 + l16];

  float localN = 0.f, localC = 0.f;
#pragma unroll
  for (int r = 0; r < 8; ++r) {
    const int brow = w * 16 + r + 8 * hi;  // batch index
    const int tgt = E[(long)brow * 512 + t + 1];
    float vals[8];
    float m = -INFINITY;
#pragma unroll
    for (int j = 0; j < 8; ++j) {
      vals[j] = acc[j][r] + bcol[j];
      m = fmaxf(m, vals[j]);
    }
#pragma unroll
    for (int s = 1; s < 16; s <<= 1) m = fmaxf(m, __shfl_xor(m, s, 32));
    float se = 0.f;
#pragma unroll
    for (int j = 0; j < 8; ++j) se += __expf(vals[j] - m);
#pragma unroll
    for (int s = 1; s < 16; s <<= 1) se += __shfl_xor(se, s, 32);
    const float logZ = m + __logf(se);
    if (l16 == (tgt & 15) && tgt != 0) {  // PAD_IDX == 0
      localN += logZ - vals[tgt >> 4];
      localC += 1.f;
    }
  }
  red[tid] = localN;
  __syncthreads();
  for (int s = 128; s > 0; s >>= 1) {
    if (tid < s) red[tid] += red[tid + s];
    __syncthreads();
  }
  if (tid == 0) nll_out[t] = red[0];
  __syncthreads();
  red[tid] = localC;
  __syncthreads();
  for (int s = 128; s > 0; s >>= 1) {
    if (tid < s) red[tid] += red[tid + s];
    __syncthreads();
  }
  if (tid == 0) cnt_out[t] = red[0];
}

// ---------------- helpers ----------------------------------------------------
__global__ void cvt2d_kernel(const float* __restrict__ src, __bf16* __restrict__ dst,
                             int rows, int cols, int dstld, int dstoff) {
  const long n = (long)rows * cols;
  for (long i = (long)blockIdx.x * blockDim.x + threadIdx.x; i < n;
       i += (long)gridDim.x * blockDim.x) {
    const int r = (int)(i / cols), c = (int)(i % cols);
    dst[(long)r * dstld + dstoff + c] = (__bf16)src[i];
  }
}

__global__ void zero_kernel(unsigned int* __restrict__ p, long words) {
  for (long i = (long)blockIdx.x * blockDim.x + threadIdx.x; i < words;
       i += (long)gridDim.x * blockDim.x)
    p[i] = 0u;
}

__global__ void finalize_kernel(const float* __restrict__ nll, const float* __restrict__ cnt,
                                int n, float* __restrict__ out) {
  __shared__ float r1[256], r2[256];
  const int tid = threadIdx.x;
  float a = 0.f, c = 0.f;
  for (int i = tid; i < n; i += 256) {
    a += nll[i];
    c += cnt[i];
  }
  r1[tid] = a;
  r2[tid] = c;
  __syncthreads();
  for (int s = 128; s > 0; s >>= 1) {
    if (tid < s) {
      r1[tid] += r1[tid + s];
      r2[tid] += r2[tid + s];
    }
    __syncthreads();
  }
  if (tid == 0) out[0] = r1[0] / fmaxf(r2[0], 1.f);
}

// ---------------- host driver ------------------------------------------------
extern "C" void kernel_launch(void* const* d_in, const int* in_sizes, int n_in,
                              void* d_out, int out_size, void* d_ws, size_t ws_size,
                              hipStream_t stream) {
  (void)in_sizes; (void)n_in; (void)out_size; (void)ws_size;
  enum { Bb = 128, Lc = 512, Le = 512, H = 512, V = 128 };

  const float* C       = (const float*)d_in[0];
  const int*   C_pad   = (const int*)d_in[1];
  const int*   E       = (const int*)d_in[2];
  const float* E_emb   = (const float*)d_in[3];
  const float* enc_Wih = (const float*)d_in[4];
  const float* enc_Whh = (const float*)d_in[5];
  const float* enc_bih = (const float*)d_in[6];
  const float* enc_bhh = (const float*)d_in[7];
  const float* dec_Wih = (const float*)d_in[8];
  const float* dec_Whh = (const float*)d_in[9];
  const float* dec_bih = (const float*)d_in[10];
  const float* dec_bhh = (const float*)d_in[11];
  const float* att_W   = (const float*)d_in[12];
  const float* out_W   = (const float*)d_in[13];
  const float* out_b   = (const float*)d_in[14];
  const float* voc_W   = (const float*)d_in[15];
  const float* voc_b   = (const float*)d_in[16];

  char* ws = (char*)d_ws;
  size_t off = 0;
  auto alloc = [&](size_t bytes) {
    size_t o = off;
    off += (bytes + 255) & ~(size_t)255;
    return o;
  };

  const int KE = V + H;      // 640  (encoder combined K: x | h)
  const int KD = V + H + H;  // 1152 (decoder combined K: e | Vprev | h)

  __bf16* wEnc = (__bf16*)(ws + alloc((size_t)4 * H * KE * 2));
  __bf16* wDec = (__bf16*)(ws + alloc((size_t)4 * H * KD * 2));
  __bf16* wAtt = (__bf16*)(ws + alloc((size_t)H * H * 2));
  __bf16* wOut = (__bf16*)(ws + alloc((size_t)H * 2 * H * 2));
  __bf16* wVoc = (__bf16*)(ws + alloc((size_t)V * H * 2));
  __bf16* ehs  = (__bf16*)(ws + alloc((size_t)Bb * Lc * H * 2));  // 64 MB, L2-resident
  __bf16* hb0  = (__bf16*)(ws + alloc((size_t)Bb * H * 2));
  __bf16* hb1  = (__bf16*)(ws + alloc((size_t)Bb * H * 2));
  float*  cb   = (float*)(ws + alloc((size_t)Bb * H * 4));
  __bf16* vprev= (__bf16*)(ws + alloc((size_t)Bb * H * 2));
  float*  qb   = (float*)(ws + alloc((size_t)Bb * H * 4));
  __bf16* hattn= (__bf16*)(ws + alloc((size_t)Bb * 2 * H * 2));
  __bf16* tanhV= (__bf16*)(ws + alloc((size_t)Bb * H * 2));
  float*  nllp = (float*)(ws + alloc((size_t)(Le - 1) * 4));
  float*  cntp = (float*)(ws + alloc((size_t)(Le - 1) * 4));
  __bf16* hb[2] = {hb0, hb1};

  // ---- pack weights to bf16 (once per call; deterministic) ----
  cvt2d_kernel<<<256, 256, 0, stream>>>(enc_Wih, wEnc, 4 * H, V, KE, 0);
  cvt2d_kernel<<<256, 256, 0, stream>>>(enc_Whh, wEnc, 4 * H, H, KE, V);
  cvt2d_kernel<<<256, 256, 0, stream>>>(dec_Wih, wDec, 4 * H, V + H, KD, 0);
  cvt2d_kernel<<<256, 256, 0, stream>>>(dec_Whh, wDec, 4 * H, H, KD, V + H);
  cvt2d_kernel<<<256, 256, 0, stream>>>(att_W, wAtt, H, H, H, 0);
  cvt2d_kernel<<<256, 256, 0, stream>>>(out_W, wOut, H, 2 * H, 2 * H, 0);
  cvt2d_kernel<<<256, 256, 0, stream>>>(voc_W, wVoc, V, H, H, 0);

  // ---- init state ----
  zero_kernel<<<64, 256, 0, stream>>>((unsigned int*)hb0, (long)Bb * H / 2);
  zero_kernel<<<64, 256, 0, stream>>>((unsigned int*)cb, (long)Bb * H);
  zero_kernel<<<64, 256, 0, stream>>>((unsigned int*)vprev, (long)Bb * H / 2);

  // ---- encoder: 512 fused LSTM steps, h written to enc_hs (bf16) ----
  for (int t = 0; t < Lc; ++t) {
    lstm_step_kernel<<<16, 256, 0, stream>>>(
        C + (long)t * V, (long)Lc * V, V,
        (const __bf16*)nullptr, 0,
        hb[t & 1], H,
        KE, wEnc, enc_bih, enc_bhh,
        cb, hb[(t + 1) & 1],
        ehs + (long)t * H, (long)Lc * H);
  }
  // final h lands in hb[0] (512 steps), c stays in cb -> decoder initial state.

  // ---- decoder: 511 steps ----
  for (int t = 0; t < Le - 1; ++t) {
    __bf16* hin = hb[t & 1];
    __bf16* hout = hb[(t + 1) & 1];
    lstm_step_kernel<<<16, 256, 0, stream>>>(
        E_emb + (long)t * V, (long)Le * V, V,
        vprev, H,
        hin, H,
        KD, wDec, dec_bih, dec_bhh,
        cb, hout,
        (__bf16*)nullptr, 0);
    // q = h @ att_W^T  (f32 out)
    gemm_kernel<<<16, 256, 0, stream>>>(hout, H, wAtt, H, nullptr, qb, nullptr, nullptr);
    // scores -> masked softmax -> context; builds hattn = [h | attn] (bf16)
    attn_kernel<<<Bb, 256, 0, stream>>>(ehs, qb, C_pad, hout, hattn);
    // Vnew = hattn @ out_W^T + out_b ; raw bf16 -> vprev, tanh bf16 -> vocab input
    gemm_kernel<<<16, 256, 0, stream>>>(hattn, 2 * H, wOut, H, out_b, nullptr, vprev, tanhV);
    // logits + log-softmax + masked NLL (deterministic per-step partials)
    vocab_ce_kernel<<<1, 256, 0, stream>>>(tanhV, wVoc, voc_b, E, t, nllp, cntp);
  }

  finalize_kernel<<<1, 256, 0, stream>>>(nllp, cntp, Le - 1, (float*)d_out);
}
